// OfflineSlidingWindowAttn_35837207118659
// MI455X (gfx1250) — matrix operand order repaired and checked by reading, not compiled, not run on hardware
//
#include <hip/hip_runtime.h>
#include <hip/hip_bf16.h>
#include <math.h>

#define HEAD_DIM    128
#define NUM_Q_HEAD  16
#define NUM_KV_HEAD 4
#define WINDOW      1024
#define CAP_LOG2E   43.280851226668961f   /* 30 * log2(e): exp(30*tanh) = exp2(CAP_LOG2E*tanh) */
#define CLIP_SCALE  1.06f
#define CLIP_OFF    (-0.03f)
#define SCALE       0.088388347648318447f /* 1/sqrt(128) */

#define KSTR 136   /* K tile LDS row stride in halves (128 + 8 pad) */
#define VSTR 40    /* Vt LDS row stride in halves (32 + 8 pad)      */
#define PSTR 40    /* P  LDS row stride in halves (32 + 8 pad)      */

typedef _Float16 v16h __attribute__((ext_vector_type(16)));
typedef _Float16 v8h  __attribute__((ext_vector_type(8)));
typedef float    v8f  __attribute__((ext_vector_type(8)));

__device__ __forceinline__ float fast_tanh(float x) {
#if __has_builtin(__builtin_amdgcn_tanhf)
  return __builtin_amdgcn_tanhf(x);
#elif __has_builtin(__builtin_amdgcn_tanh_f32)
  return __builtin_amdgcn_tanh_f32(x);
#else
  return tanhf(x);
#endif
}

__device__ __forceinline__ float fast_exp2(float x) {
#if __has_builtin(__builtin_amdgcn_exp2f)
  return __builtin_amdgcn_exp2f(x);
#else
  return exp2f(x);
#endif
}

__device__ __forceinline__ v16h mk16(v8h lo, v8h hi) {
  v16h r;
#pragma unroll
  for (int i = 0; i < 8; ++i) { r[i] = lo[i]; r[i + 8] = hi[i]; }
  return r;
}

__device__ __forceinline__ v16h lds16(const _Float16* p) {
  v8h lo = *(const v8h*)p;
  v8h hi = *(const v8h*)(p + 8);
  return mk16(lo, hi);
}

__device__ __forceinline__ v8h cvt8_scaled(const float* p) {
  v8h r;
#pragma unroll
  for (int i = 0; i < 8; ++i) r[i] = (_Float16)(p[i] * SCALE);
  return r;
}

__device__ __forceinline__ void load8x4(const float* p, float4* r) {
#pragma unroll
  for (int jj = 0; jj < 8; ++jj) r[jj] = ((const float4*)p)[jj];
}

/* convert 32 f32 (in regs) -> 32 f16, row-major into K tile */
__device__ __forceinline__ void stage_k_regs(const float4* f, _Float16* dst) {
#pragma unroll
  for (int jj = 0; jj < 4; ++jj) {
    float4 f0 = f[2 * jj], f1 = f[2 * jj + 1];
    v8h hh;
    hh[0] = (_Float16)f0.x; hh[1] = (_Float16)f0.y;
    hh[2] = (_Float16)f0.z; hh[3] = (_Float16)f0.w;
    hh[4] = (_Float16)f1.x; hh[5] = (_Float16)f1.y;
    hh[6] = (_Float16)f1.z; hh[7] = (_Float16)f1.w;
    *(v8h*)(dst + jj * 8) = hh;
  }
}

/* convert 32 f32 (in regs) -> 32 f16, transposed into Vt tile */
__device__ __forceinline__ void stage_v_regs(const float4* f, _Float16* vt,
                                             int srow, int scb) {
#pragma unroll
  for (int jj = 0; jj < 8; ++jj) {
    float4 fv = f[jj];
    vt[(scb + jj * 4 + 0) * VSTR + srow] = (_Float16)fv.x;
    vt[(scb + jj * 4 + 1) * VSTR + srow] = (_Float16)fv.y;
    vt[(scb + jj * 4 + 2) * VSTR + srow] = (_Float16)fv.z;
    vt[(scb + jj * 4 + 3) * VSTR + srow] = (_Float16)fv.w;
  }
}

__global__ __launch_bounds__(128, 1)
void OfflineSlidingWindowAttn_35837207118659_kernel(
    const float* __restrict__ q, const float* __restrict__ k,
    const float* __restrict__ v, float* __restrict__ out, int S) {
  __shared__ __align__(16) _Float16 Ksh[2][32 * KSTR];        /* double-buffered K tiles  */
  __shared__ __align__(16) _Float16 Vtsh[2][HEAD_DIM * VSTR]; /* double-buffered Vt tiles */
  __shared__ __align__(16) _Float16 Psh[4 * 16 * PSTR];       /* per-wave 16x32 probs     */

  const int tid  = threadIdx.x;
  const int lane = tid & 31;
  const int wv   = tid >> 5;
  const int nIdx = lane & 15;   /* N / M-low index  */
  const int hi   = lane >> 4;   /* lane-half select */

  const int nQB = S >> 6;
  const int bid = blockIdx.x;
  const int qb  = bid % nQB;
  const int h   = (bid / nQB) % NUM_Q_HEAD;
  const int b   = bid / (nQB * NUM_Q_HEAD);
  const int hkv = h / (NUM_Q_HEAD / NUM_KV_HEAD);

  const int q0 = qb * 64;
  const int qw = q0 + wv * 16;        /* this wave's first query row */

  /* ---- Q A-fragments (scale folded in), 16x32 per K-chunk, 4 chunks ---- */
  const float* qrow =
      q + ((size_t)((size_t)b * S + (qw + nIdx)) * NUM_Q_HEAD + h) * HEAD_DIM;
  v16h aq[4];
#pragma unroll
  for (int c = 0; c < 4; ++c) {
    v8h lo = cvt8_scaled(qrow + c * 32 + hi * 16);
    v8h hh = cvt8_scaled(qrow + c * 32 + 8 + hi * 16);
    aq[c] = mk16(lo, hh);
  }

  /* tanh cap bounds scores to [-30,30]: exp(s) and its sum stay in fp32
     range, so no running-max is needed at all. */
  float lrow[8];
#pragma unroll
  for (int r = 0; r < 8; ++r) lrow[r] = 0.0f;

  int kt_lo = q0 - WINDOW; if (kt_lo < 0) kt_lo = 0;
  const int kt_end = q0 + 64;

  const int srow = tid >> 2;        /* staging: key row 0..31        */
  const int scb  = (tid & 3) * 32;  /* staging: dim chunk 0/32/64/96 */

  const size_t tstep = (size_t)32 * NUM_KV_HEAD * HEAD_DIM;  /* advance 32 keys */
  const float* kstage0 =
      k + ((size_t)((size_t)b * S + (kt_lo + srow)) * NUM_KV_HEAD + hkv) * HEAD_DIM + scb;
  const float* vstage0 =
      v + ((size_t)((size_t)b * S + (kt_lo + srow)) * NUM_KV_HEAD + hkv) * HEAD_DIM + scb;

  /* ================= Pass A: softmax denominator ================= */
  {
    const float* kstage = kstage0;
    float4 kr[8];
    load8x4(kstage, kr);
    stage_k_regs(kr, &Ksh[0][srow * KSTR + scb]);   /* stage tile 0 */
    int par = 0;
    for (int kt = kt_lo; kt < kt_end; kt += 32, kstage += tstep) {
      __syncthreads();                               /* tile kt visible in Ksh[par] */
      const bool more = (kt + 32 < kt_end);
      if (more) {
        load8x4(kstage + tstep, kr);                 /* overlap VMEM with compute */
        if (kt + 64 < kt_end) __builtin_prefetch(kstage + 2 * tstep, 0, 0);
      }

      const int dq = qw + 8 * hi - kt;               /* qi - key = dq + r - keyoff */
#pragma unroll
      for (int sub = 0; sub < 2; ++sub) {
        v8f Sacc;
#pragma unroll
        for (int r = 0; r < 8; ++r) Sacc[r] = 0.0f;
#pragma unroll
        for (int c = 0; c < 4; ++c) {
          v16h bf = lds16(&Ksh[par][(sub * 16 + nIdx) * KSTR + c * 32 + hi * 16]);
          Sacc = __builtin_amdgcn_wmma_f32_16x16x32_f16(
              false, aq[c], false, bf, (short)0, Sacc, false, false);
        }
        const int koff = sub * 16 + nIdx;
#pragma unroll
        for (int r = 0; r < 8; ++r) {
          float e = fast_exp2(CAP_LOG2E * fast_tanh(Sacc[r]));
          const bool ok = (unsigned)(dq + r - koff) <= (unsigned)WINDOW;
          e = ok ? e : 0.0f;
          e += __shfl_xor(e, 1, 32);
          e += __shfl_xor(e, 2, 32);
          e += __shfl_xor(e, 4, 32);
          e += __shfl_xor(e, 8, 32);
          lrow[r] += e;
        }
      }

      if (more) stage_k_regs(kr, &Ksh[par ^ 1][srow * KSTR + scb]);
      par ^= 1;
    }
  }

  float linv[8];
#pragma unroll
  for (int r = 0; r < 8; ++r) linv[r] = 1.0f / lrow[r];

  v8f Oacc[8];
#pragma unroll
  for (int d = 0; d < 8; ++d) {
#pragma unroll
    for (int r = 0; r < 8; ++r) Oacc[d][r] = 0.0f;
  }

  /* ================= Pass B: clipped probabilities x V ================= */
  {
    __syncthreads();   /* pass A's last computes must finish before restaging */
    const float* kstage = kstage0;
    const float* vstage = vstage0;
    float4 kr[8], vr[8];
    load8x4(kstage, kr);
    load8x4(vstage, vr);
    stage_k_regs(kr, &Ksh[0][srow * KSTR + scb]);
    stage_v_regs(vr, &Vtsh[0][0], srow, scb);
    int par = 0;
    for (int kt = kt_lo; kt < kt_end; kt += 32, kstage += tstep, vstage += tstep) {
      __syncthreads();
      const bool more = (kt + 32 < kt_end);
      if (more) {
        load8x4(kstage + tstep, kr);
        load8x4(vstage + tstep, vr);
        if (kt + 64 < kt_end) {
          __builtin_prefetch(kstage + 2 * tstep, 0, 0);
          __builtin_prefetch(vstage + 2 * tstep, 0, 0);
        }
      }

      _Float16* pw = &Psh[wv * 16 * PSTR];
      const int dq = qw + 8 * hi - kt;
#pragma unroll
      for (int sub = 0; sub < 2; ++sub) {
        v8f Sacc;
#pragma unroll
        for (int r = 0; r < 8; ++r) Sacc[r] = 0.0f;
#pragma unroll
        for (int c = 0; c < 4; ++c) {
          v16h bf = lds16(&Ksh[par][(sub * 16 + nIdx) * KSTR + c * 32 + hi * 16]);
          Sacc = __builtin_amdgcn_wmma_f32_16x16x32_f16(
              false, aq[c], false, bf, (short)0, Sacc, false, false);
        }
        const int koff = sub * 16 + nIdx;
#pragma unroll
        for (int r = 0; r < 8; ++r) {
          float e = fast_exp2(CAP_LOG2E * fast_tanh(Sacc[r]));
          const bool ok = (unsigned)(dq + r - koff) <= (unsigned)WINDOW;
          e = ok ? e : 0.0f;
          float a = fmaf(CLIP_SCALE, e * linv[r], CLIP_OFF);
          a = fminf(fmaxf(a, 0.0f), 1.0f);
          pw[(r + 8 * hi) * PSTR + sub * 16 + nIdx] = (_Float16)a;
        }
      }
      /* P buffer is wave-private; LDS executes a wave's DS ops in order. */
      __builtin_amdgcn_wave_barrier();

      const _Float16* prow = &pw[nIdx * PSTR];
      v8h pl = *(const v8h*)(prow + hi * 16);
      v8h ph = *(const v8h*)(prow + hi * 16 + 8);
      v16h ap = mk16(pl, ph);
#pragma unroll
      for (int d = 0; d < 8; ++d) {
        v16h bv = lds16(&Vtsh[par][(d * 16 + nIdx) * VSTR + hi * 16]);
        Oacc[d] = __builtin_amdgcn_wmma_f32_16x16x32_f16(
            false, ap, false, bv, (short)0, Oacc[d], false, false);
      }

      if (more) {
        stage_k_regs(kr, &Ksh[par ^ 1][srow * KSTR + scb]);
        stage_v_regs(vr, &Vtsh[par ^ 1][0], srow, scb);
      }
      par ^= 1;
    }
  }

  /* ---- store O (already normalized inside the clip) ---- */
#pragma unroll
  for (int r = 0; r < 8; ++r) {
    const int qi = qw + r + 8 * hi;
    float* orow = out + ((size_t)((size_t)b * S + qi) * NUM_Q_HEAD + h) * HEAD_DIM;
#pragma unroll
    for (int d = 0; d < 8; ++d) orow[d * 16 + nIdx] = Oacc[d][r];
  }
}

extern "C" void kernel_launch(void* const* d_in, const int* in_sizes, int n_in,
                              void* d_out, int out_size, void* d_ws, size_t ws_size,
                              hipStream_t stream) {
  const float* q = (const float*)d_in[0];
  const float* k = (const float*)d_in[1];
  const float* v = (const float*)d_in[2];
  float* out = (float*)d_out;

  const int B = 2;
  const int S = in_sizes[0] / (B * NUM_Q_HEAD * HEAD_DIM);   /* 2048 */

  dim3 grid(B * NUM_Q_HEAD * (S / 64));
  dim3 block(128);
  OfflineSlidingWindowAttn_35837207118659_kernel<<<grid, block, 0, stream>>>(q, k, v, out, S);
}
